// AnimalT_67156108640707
// MI455X (gfx1250) — compile-verified, matrix operand order checked by reading, not run on hardware
//
#include <hip/hip_runtime.h>
#include <math.h>

// ---------------------------------------------------------------------------
// Types for CDNA5 WMMA (wave32, 16x16x32 bf16 -> f32)
// ---------------------------------------------------------------------------
typedef __bf16 bf16;
typedef __attribute__((ext_vector_type(16))) __bf16 v16bf;
typedef __attribute__((ext_vector_type(8)))  __bf16 v8bf;
typedef __attribute__((ext_vector_type(8)))  float  v8f;

// Model dims (match reference)
#define BB   4
#define TT   1024
#define CC   1024
#define HH   16
#define DH   64
#define LL   8
#define DIN  128
#define DOUT 256
#define FF   4096
#define MROW (BB * TT)   // 4096

// GEMM epilogue flags
#define EF_BIAS  1
#define EF_OBF16 2
#define EF_RES   4
#define EF_GELU  8
#define EF_TANH  16
#define EF_DELTA 32

// LDS tile row stride (elements): 32 payload + 8 pad -> 80B rows, bank-conflict
// free for 16B ds_load_b128 fragment reads (16 disjoint 4-bank spans).
#define LROW 40

struct GemmP {
  const bf16*  A;      // M x K row-major (lda)
  const bf16*  B;      // N x K row-major (ldb)  == transposed weight
  void*        out;    // f32 or bf16 (ldc)
  const float* bias;   // [N]
  const float* res;    // residual f32, same indexing as f32 out
  const float* xnext;  // tokens, for DELTA epilogue
  int M, N, K, lda, ldb, ldc;
  long long sA, sB, sC;   // per-blockIdx.z element strides
  int Tdim;               // for DELTA
};

// generic -> LDS byte offset (addrspace(3) pointers are 32-bit offsets)
__device__ __forceinline__ unsigned lds_off_of(const void* p) {
  return (unsigned)(size_t)(__attribute__((address_space(3))) const void*)p;
}

// one 16B async chunk: LDS[lds_off] = MEM[gaddr]  (ASYNCcnt-tracked)
__device__ __forceinline__ void async_cp16(unsigned lds_off, const void* gaddr) {
  asm volatile("global_load_async_to_lds_b128 %0, %1, off"
               :: "v"(lds_off), "v"((unsigned long long)(size_t)gaddr)
               : "memory");
}

// Stage one 128x32 A tile and one 128x32 B tile into LDS (async).
// 512 chunks per tile, 128 threads -> 4 passes, 2 async ops per pass.
__device__ __forceinline__ void stage_tile(
    const bf16* gA, const bf16* gB, int lda, int ldb,
    int mmax, int nmax, bf16* sA, bf16* sB, int tid, int k0) {
#pragma unroll
  for (int pass = 0; pass < 4; ++pass) {
    const int idx = pass * 128 + tid;
    const int row = idx >> 2;
    const int c   = (idx & 3) * 8;                 // bf16 elements within row
    const int ra  = row < mmax ? row : mmax - 1;   // clamp (no divergence)
    const int rb  = row < nmax ? row : nmax - 1;
    async_cp16(lds_off_of(sA + row * LROW + c),
               gA + (size_t)ra * lda + k0 + c);
    async_cp16(lds_off_of(sB + row * LROW + c),
               gB + (size_t)rb * ldb + k0 + c);
  }
}

// A/B fragment from LDS tile per CDNA5 16-bit A-matrix 16x32 layout:
// lane<16: row=lane, K {0..7, 16..23}; lane>=16: row=lane-16, K {8..15, 24..31}
__device__ __forceinline__ v16bf lds_frag(const bf16* s, int row0, int lane) {
  const int half = lane >> 4;
  const int r    = lane & 15;
  const bf16* p  = s + (size_t)(row0 + r) * LROW + half * 8;
  v8bf lo = *(const v8bf*)(p);
  v8bf hi = *(const v8bf*)(p + 16);
  v16bf f;
#pragma unroll
  for (int i = 0; i < 8; ++i) { f[i] = lo[i]; f[i + 8] = hi[i]; }
  return f;
}

// ---------------------------------------------------------------------------
// WMMA GEMM: 128 threads = 4 waves; block tile 128x128x(K steps of 32).
// Double-buffered LDS tiles fed by global_load_async_to_lds_b128; each wave
// computes a 64x64 tile = 4x4 grid of v_wmma_f32_16x16x32_bf16.
// ---------------------------------------------------------------------------
template <int F>
__global__ __launch_bounds__(128)
void gemm_bf16_wmma(GemmP p) {
  __shared__ __align__(16) bf16 smem[2][2][128 * LROW];  // [buf][A/B] = 40KB

  const int z    = blockIdx.z;
  const bf16* A  = p.A + (size_t)z * p.sA;
  const bf16* Bt = p.B + (size_t)z * p.sB;

  const int tid  = threadIdx.x;
  const int lane = tid & 31;
  const int wave = tid >> 5;
  const int m_blk = blockIdx.x * 128;
  const int n_blk = blockIdx.y * 128;
  const int wm = (wave >> 1) * 64;
  const int wn = (wave & 1) * 64;
  const int mmax = min(128, p.M - m_blk);
  const int nmax = min(128, p.N - n_blk);

  const bf16* gA = A  + (size_t)m_blk * p.lda;
  const bf16* gB = Bt + (size_t)n_blk * p.ldb;

  v8f acc[4][4];
#pragma unroll
  for (int i = 0; i < 4; ++i)
#pragma unroll
    for (int j = 0; j < 4; ++j)
#pragma unroll
      for (int e = 0; e < 8; ++e) acc[i][j][e] = 0.0f;

  const int nk = p.K >> 5;
  stage_tile(gA, gB, p.lda, p.ldb, mmax, nmax, smem[0][0], smem[0][1], tid, 0);

  for (int kt = 0; kt < nk; ++kt) {
    const int cur = kt & 1;
    if (kt + 1 < nk) {
      stage_tile(gA, gB, p.lda, p.ldb, mmax, nmax,
                 smem[cur ^ 1][0], smem[cur ^ 1][1], tid, (kt + 1) * 32);
      asm volatile("s_wait_asynccnt 0x8" ::: "memory");  // tile kt done
    } else {
      asm volatile("s_wait_asynccnt 0x0" ::: "memory");  // drain
    }
    __syncthreads();   // all waves' async tiles visible

    const bf16* sA = smem[cur][0];
    const bf16* sB = smem[cur][1];
    v16bf af[4], bfr[4];
#pragma unroll
    for (int i = 0; i < 4; ++i) af[i]  = lds_frag(sA, wm + i * 16, lane);
#pragma unroll
    for (int j = 0; j < 4; ++j) bfr[j] = lds_frag(sB, wn + j * 16, lane);
#pragma unroll
    for (int i = 0; i < 4; ++i)
#pragma unroll
      for (int j = 0; j < 4; ++j)
        acc[i][j] = __builtin_amdgcn_wmma_f32_16x16x32_bf16(
            false, af[i], false, bfr[j], (short)0, acc[i][j], false, false);

    __syncthreads();   // protect buffer `cur` before it is re-staged
  }

  // Epilogue.  C/D layout: VGPR e holds M = e (lanes 0-15) / e+8 (lanes 16-31),
  // N = lane & 15.
  const int half = lane >> 4;
  const int r    = lane & 15;
  const int m_base = m_blk + wm;
  const int n_base = n_blk + wn;
  bool mv[4], nv[4];
#pragma unroll
  for (int i = 0; i < 4; ++i) {
    mv[i] = (m_base + i * 16) < p.M;
    nv[i] = (n_base + i * 16) < p.N;
  }
  float* outf = (float*)p.out + (size_t)z * ((F & EF_OBF16) ? 0 : p.sC);
  bf16*  outb = (bf16*)p.out  + (size_t)z * ((F & EF_OBF16) ? p.sC : 0);

#pragma unroll
  for (int i = 0; i < 4; ++i) {
    if (!mv[i]) continue;
#pragma unroll
    for (int j = 0; j < 4; ++j) {
      if (!nv[j]) continue;
      const int col = n_base + j * 16 + r;
#pragma unroll
      for (int e = 0; e < 8; ++e) {
        const int rr = m_base + i * 16 + half * 8 + e;
        float v = acc[i][j][e];
        if (F & EF_BIAS) v += p.bias[col];
        if (F & EF_GELU) v = 0.5f * v * (1.0f + erff(v * 0.70710678118654752f));
        if (F & EF_TANH) v = tanhf(v);
        if (F & EF_RES)  v += p.res[(size_t)rr * p.ldc + col];
        if (F & EF_DELTA) {
          const int b = rr / p.Tdim, t = rr % p.Tdim;
          if (t == p.Tdim - 1) continue;                  // pred[:, :-1]
          v -= p.xnext[((size_t)rr + 1) * p.N + col];     // - x[:, 1:]
          outf[((size_t)b * (p.Tdim - 1) + t) * p.N + col] = v;
        } else if (F & EF_OBF16) {
          outb[(size_t)rr * p.ldc + col] = (bf16)v;
        } else {
          outf[(size_t)rr * p.ldc + col] = v;
        }
      }
    }
  }
}

// ---------------------------------------------------------------------------
// LayerNorm (f32 in -> bf16 out), one block per row
// ---------------------------------------------------------------------------
__global__ __launch_bounds__(256)
void ln_bf16_kernel(const float* __restrict__ x, const float* __restrict__ g,
                    const float* __restrict__ b, bf16* __restrict__ out, int W) {
  const int row = blockIdx.x;
  const float* xr = x + (size_t)row * W;
  float s = 0.f, s2 = 0.f;
  for (int i = threadIdx.x; i < W; i += blockDim.x) {
    float v = xr[i]; s += v; s2 += v * v;
  }
  __shared__ float sh0[8], sh1[8];
#pragma unroll
  for (int o = 16; o > 0; o >>= 1) {
    s  += __shfl_xor(s, o, 32);
    s2 += __shfl_xor(s2, o, 32);
  }
  if ((threadIdx.x & 31) == 0) { sh0[threadIdx.x >> 5] = s; sh1[threadIdx.x >> 5] = s2; }
  __syncthreads();
  const int nw = blockDim.x >> 5;
  s = 0.f; s2 = 0.f;
  for (int w = 0; w < nw; ++w) { s += sh0[w]; s2 += sh1[w]; }
  const float mu   = s / W;
  const float var  = s2 / W - mu * mu;
  const float rstd = rsqrtf(var + 1e-5f);
  for (int i = threadIdx.x; i < W; i += blockDim.x)
    out[(size_t)row * W + i] = (bf16)((xr[i] - mu) * rstd * g[i] + b[i]);
}

// ---------------------------------------------------------------------------
// Causal softmax over S (H,T,T) f32 -> P (H,T,T) bf16; scale folded in here
// ---------------------------------------------------------------------------
__global__ __launch_bounds__(256)
void softmax_causal(const float* __restrict__ S, bf16* __restrict__ P,
                    int T, float scale) {
  const int q = blockIdx.x, h = blockIdx.y;
  const float* sr = S + ((size_t)h * T + q) * T;
  bf16*        pr = P + ((size_t)h * T + q) * T;
  const int nvalid = q + 1;
  __shared__ float sh[8];
  const int nw = blockDim.x >> 5;

  float mx = -3.4e38f;
  for (int i = threadIdx.x; i < nvalid; i += blockDim.x)
    mx = fmaxf(mx, sr[i] * scale);
#pragma unroll
  for (int o = 16; o > 0; o >>= 1) mx = fmaxf(mx, __shfl_xor(mx, o, 32));
  if ((threadIdx.x & 31) == 0) sh[threadIdx.x >> 5] = mx;
  __syncthreads();
  mx = -3.4e38f;
  for (int w = 0; w < nw; ++w) mx = fmaxf(mx, sh[w]);
  __syncthreads();

  float sum = 0.f;
  for (int i = threadIdx.x; i < nvalid; i += blockDim.x)
    sum += __expf(sr[i] * scale - mx);
#pragma unroll
  for (int o = 16; o > 0; o >>= 1) sum += __shfl_xor(sum, o, 32);
  if ((threadIdx.x & 31) == 0) sh[threadIdx.x >> 5] = sum;
  __syncthreads();
  float tot = 0.f;
  for (int w = 0; w < nw; ++w) tot += sh[w];
  const float inv = 1.0f / tot;
  for (int i = threadIdx.x; i < T; i += blockDim.x)
    pr[i] = (i < nvalid) ? (bf16)(__expf(sr[i] * scale - mx) * inv) : (bf16)0.0f;
}

// ---------------------------------------------------------------------------
// Weight transpose + f32->bf16:  in[l][k][n] -> out[l][n][k]
// ---------------------------------------------------------------------------
__global__ void wcvt_kernel(const float* __restrict__ in, bf16* __restrict__ out,
                            int L, int K, int N) {
  size_t idx = (size_t)blockIdx.x * blockDim.x + threadIdx.x;
  size_t tot = (size_t)L * K * N;
  if (idx >= tot) return;
  int n = (int)(idx % N);
  size_t t = idx / N;
  int k = (int)(t % K);
  int l = (int)(t / K);
  out[((size_t)l * N + n) * K + k] = (bf16)in[idx];
}

__global__ void cvt_bf16_kernel(const float* __restrict__ in,
                                bf16* __restrict__ out, size_t n) {
  size_t idx = (size_t)blockIdx.x * blockDim.x + threadIdx.x;
  if (idx < n) out[idx] = (bf16)in[idx];
}

// V (B,T,C) bf16 -> Vt (B,H,DH,T) bf16
__global__ void vtrans_kernel(const bf16* __restrict__ v, bf16* __restrict__ vt) {
  size_t idx = (size_t)blockIdx.x * blockDim.x + threadIdx.x;
  size_t tot = (size_t)BB * HH * DH * TT;
  if (idx >= tot) return;
  int t = (int)(idx % TT);
  size_t r = idx / TT;
  int d = (int)(r % DH); r /= DH;
  int h = (int)(r % HH);
  int b = (int)(r / HH);
  vt[idx] = v[((size_t)b * TT + t) * CC + h * DH + d];
}

// ---------------------------------------------------------------------------
// Host-side helpers
// ---------------------------------------------------------------------------
template <int F>
static void gemm(hipStream_t s, GemmP p, int Z) {
  dim3 grid((p.M + 127) / 128, (p.N + 127) / 128, Z);
  hipLaunchKernelGGL((gemm_bf16_wmma<F>), grid, dim3(128), 0, s, p);
}

static GemmP mk(const bf16* A, const bf16* B, void* out, const float* bias,
                const float* res, const float* xnext, int M, int N, int K,
                int lda, int ldb, int ldc, long long sA, long long sB,
                long long sC, int Tdim) {
  GemmP p; p.A = A; p.B = B; p.out = out; p.bias = bias; p.res = res;
  p.xnext = xnext; p.M = M; p.N = N; p.K = K; p.lda = lda; p.ldb = ldb;
  p.ldc = ldc; p.sA = sA; p.sB = sB; p.sC = sC; p.Tdim = Tdim;
  return p;
}

extern "C" void kernel_launch(void* const* d_in, const int* in_sizes, int n_in,
                              void* d_out, int out_size, void* d_ws, size_t ws_size,
                              hipStream_t stream) {
  (void)in_sizes; (void)n_in; (void)out_size; (void)ws_size;
  const float* tokens  = (const float*)d_in[0];
  const float* emb_w   = (const float*)d_in[1];
  const float* emb_b   = (const float*)d_in[2];
  const float* ln1_g   = (const float*)d_in[3];
  const float* ln1_b   = (const float*)d_in[4];
  const float* Wq      = (const float*)d_in[5];
  const float* bq      = (const float*)d_in[6];
  const float* Wk      = (const float*)d_in[7];
  const float* bk      = (const float*)d_in[8];
  const float* Wv      = (const float*)d_in[9];
  const float* bv      = (const float*)d_in[10];
  const float* Wo      = (const float*)d_in[11];
  const float* bo      = (const float*)d_in[12];
  const float* ln2_g   = (const float*)d_in[13];
  const float* ln2_b   = (const float*)d_in[14];
  const float* W1      = (const float*)d_in[15];
  const float* b1      = (const float*)d_in[16];
  const float* W2      = (const float*)d_in[17];
  const float* b2      = (const float*)d_in[18];
  const float* lnf_g   = (const float*)d_in[19];
  const float* lnf_b   = (const float*)d_in[20];
  const float* proj_w  = (const float*)d_in[21];
  const float* proj_b  = (const float*)d_in[22];
  const float* dec_w1  = (const float*)d_in[23];
  const float* dec_b1  = (const float*)d_in[24];
  const float* dec_lng = (const float*)d_in[25];
  const float* dec_lnb = (const float*)d_in[26];
  const float* dec_w2  = (const float*)d_in[27];
  const float* dec_b2  = (const float*)d_in[28];
  float* out = (float*)d_out;

  // Workspace sub-allocation (256B aligned)
  char* base = (char*)d_ws;
  size_t off = 0;
  auto alloc = [&](size_t bytes) -> char* {
    char* r = base + off;
    off = (off + bytes + 255) & ~(size_t)255;
    return r;
  };
  bf16* embT  = (bf16*)alloc((size_t)DIN * CC * 2);
  bf16* WqT   = (bf16*)alloc((size_t)LL * CC * CC * 2);
  bf16* WkT   = (bf16*)alloc((size_t)LL * CC * CC * 2);
  bf16* WvT   = (bf16*)alloc((size_t)LL * CC * CC * 2);
  bf16* WoT   = (bf16*)alloc((size_t)LL * CC * CC * 2);
  bf16* W1T   = (bf16*)alloc((size_t)LL * CC * FF * 2);
  bf16* W2T   = (bf16*)alloc((size_t)LL * FF * CC * 2);
  bf16* projT = (bf16*)alloc((size_t)CC * DOUT * 2);
  bf16* dec1T = (bf16*)alloc((size_t)DOUT * DOUT * 2);
  bf16* dec2T = (bf16*)alloc((size_t)DOUT * DIN * 2);
  bf16* xbf   = (bf16*)alloc((size_t)MROW * DIN * 2);
  float* h    = (float*)alloc((size_t)MROW * CC * 4);
  bf16* abf   = (bf16*)alloc((size_t)MROW * CC * 2);
  bf16* q     = (bf16*)alloc((size_t)MROW * CC * 2);
  bf16* k     = (bf16*)alloc((size_t)MROW * CC * 2);
  bf16* v     = (bf16*)alloc((size_t)MROW * CC * 2);
  bf16* vt    = (bf16*)alloc((size_t)BB * HH * DH * TT * 2);
  bf16* y     = (bf16*)alloc((size_t)MROW * CC * 2);
  bf16* gbf   = (bf16*)alloc((size_t)MROW * FF * 2);
  float* S    = (float*)alloc((size_t)HH * TT * TT * 4);   // per-batch scores
  bf16* P     = (bf16*)alloc((size_t)HH * TT * TT * 2);
  bf16* fbf   = (bf16*)alloc((size_t)MROW * DOUT * 2);
  float* tf   = (float*)alloc((size_t)MROW * DOUT * 4);
  bf16* dbf   = (bf16*)alloc((size_t)MROW * DOUT * 2);

  auto wcvt = [&](const float* in, bf16* o, int L, int K, int N) {
    size_t tot = (size_t)L * K * N;
    hipLaunchKernelGGL(wcvt_kernel, dim3((unsigned)((tot + 255) / 256)),
                       dim3(256), 0, stream, in, o, L, K, N);
  };
  // 1) Convert + transpose all weights to bf16 (N x K)
  wcvt(emb_w, embT, 1, DIN, CC);
  wcvt(Wq, WqT, LL, CC, CC);
  wcvt(Wk, WkT, LL, CC, CC);
  wcvt(Wv, WvT, LL, CC, CC);
  wcvt(Wo, WoT, LL, CC, CC);
  wcvt(W1, W1T, LL, CC, FF);
  wcvt(W2, W2T, LL, FF, CC);
  wcvt(proj_w, projT, 1, CC, DOUT);
  wcvt(dec_w1, dec1T, 1, DOUT, DOUT);
  wcvt(dec_w2, dec2T, 1, DOUT, DIN);
  {
    size_t n = (size_t)MROW * DIN;
    hipLaunchKernelGGL(cvt_bf16_kernel, dim3((unsigned)((n + 255) / 256)),
                       dim3(256), 0, stream, tokens, xbf, n);
  }

  // 2) Token embedding: h = x @ emb + b   (f32 residual stream)
  gemm<EF_BIAS>(stream,
      mk(xbf, embT, h, emb_b, nullptr, nullptr,
         MROW, CC, DIN, DIN, DIN, CC, 0, 0, 0, 0), 1);

  const float scale = 1.0f / 8.0f;  // 1/sqrt(64)

  for (int l = 0; l < LL; ++l) {
    const size_t wof = (size_t)l * CC * CC;
    // ln1 -> bf16
    hipLaunchKernelGGL(ln_bf16_kernel, dim3(MROW), dim3(256), 0, stream,
                       h, ln1_g + (size_t)l * CC, ln1_b + (size_t)l * CC, abf, CC);
    // q,k,v = a @ W + b  (bf16 out)
    gemm<EF_BIAS | EF_OBF16>(stream,
        mk(abf, WqT + wof, q, bq + (size_t)l * CC, nullptr, nullptr,
           MROW, CC, CC, CC, CC, CC, 0, 0, 0, 0), 1);
    gemm<EF_BIAS | EF_OBF16>(stream,
        mk(abf, WkT + wof, k, bk + (size_t)l * CC, nullptr, nullptr,
           MROW, CC, CC, CC, CC, CC, 0, 0, 0, 0), 1);
    gemm<EF_BIAS | EF_OBF16>(stream,
        mk(abf, WvT + wof, v, bv + (size_t)l * CC, nullptr, nullptr,
           MROW, CC, CC, CC, CC, CC, 0, 0, 0, 0), 1);
    // V transpose -> (B,H,DH,T)
    {
      size_t tot = (size_t)BB * HH * DH * TT;
      hipLaunchKernelGGL(vtrans_kernel, dim3((unsigned)((tot + 255) / 256)),
                         dim3(256), 0, stream, v, vt);
    }
    // attention per batch; all H heads via grid.z
    for (int b = 0; b < BB; ++b) {
      const bf16* qb = q + (size_t)b * TT * CC;
      const bf16* kb = k + (size_t)b * TT * CC;
      // S = Q @ K^T   (per head: A = Q_bh (T x 64, lda=C), Bt = K_bh)
      gemm<0>(stream,
          mk(qb, kb, S, nullptr, nullptr, nullptr,
             TT, TT, DH, CC, CC, TT, DH, DH, (long long)TT * TT, 0), HH);
      // causal softmax (scale folded) -> P bf16
      hipLaunchKernelGGL(softmax_causal, dim3(TT, HH), dim3(256), 0, stream,
                         S, P, TT, scale);
      // y_bh = P @ V   (Bt = Vt_bh (64 x T)); write into y at col h*64
      gemm<EF_OBF16>(stream,
          mk(P, vt + (size_t)b * HH * DH * TT, y + (size_t)b * TT * CC,
             nullptr, nullptr, nullptr,
             TT, DH, TT, TT, TT, CC,
             (long long)TT * TT, (long long)DH * TT, DH, 0), HH);
    }
    // h = h + y @ Wo + bo
    gemm<EF_BIAS | EF_RES>(stream,
        mk(y, WoT + wof, h, bo + (size_t)l * CC, h, nullptr,
           MROW, CC, CC, CC, CC, CC, 0, 0, 0, 0), 1);
    // ln2 -> bf16
    hipLaunchKernelGGL(ln_bf16_kernel, dim3(MROW), dim3(256), 0, stream,
                       h, ln2_g + (size_t)l * CC, ln2_b + (size_t)l * CC, abf, CC);
    // g = gelu(a @ W1 + b1)  (bf16)
    gemm<EF_BIAS | EF_GELU | EF_OBF16>(stream,
        mk(abf, W1T + (size_t)l * CC * FF, gbf, b1 + (size_t)l * FF,
           nullptr, nullptr, MROW, FF, CC, CC, CC, FF, 0, 0, 0, 0), 1);
    // h = h + g @ W2 + b2
    gemm<EF_BIAS | EF_RES>(stream,
        mk(gbf, W2T + (size_t)l * FF * CC, h, b2 + (size_t)l * CC, h, nullptr,
           MROW, CC, FF, FF, FF, CC, 0, 0, 0, 0), 1);
  }

  // Final LN -> proj -> decoder
  hipLaunchKernelGGL(ln_bf16_kernel, dim3(MROW), dim3(256), 0, stream,
                     h, lnf_g, lnf_b, abf, CC);
  gemm<EF_BIAS | EF_OBF16>(stream,
      mk(abf, projT, fbf, proj_b, nullptr, nullptr,
         MROW, DOUT, CC, CC, CC, DOUT, 0, 0, 0, 0), 1);
  gemm<EF_BIAS | EF_TANH>(stream,
      mk(fbf, dec1T, tf, dec_b1, nullptr, nullptr,
         MROW, DOUT, DOUT, DOUT, DOUT, DOUT, 0, 0, 0, 0), 1);
  hipLaunchKernelGGL(ln_bf16_kernel, dim3(MROW), dim3(256), 0, stream,
                     tf, dec_lng, dec_lnb, dbf, DOUT);
  // pred = dln @ dec_w2 + b2 ; out = pred[:, :-1] - x[:, 1:]
  gemm<EF_BIAS | EF_DELTA>(stream,
      mk(dbf, dec2T, out, dec_b2, nullptr, tokens,
         MROW, DIN, DOUT, DOUT, DOUT, DIN, 0, 0, 0, TT), 1);
}
